// MultiHeadAttention_18425409700348
// MI455X (gfx1250) — matrix operand order repaired and checked by reading, not compiled, not run on hardware
//
#include <hip/hip_runtime.h>

typedef __attribute__((ext_vector_type(16))) _Float16 v16h;
typedef __attribute__((ext_vector_type(2)))  _Float16 v2h;
typedef __attribute__((ext_vector_type(8)))  float    v8f;

#define SEQ   2048
#define EMB   128
#define HDIM  512      // H*D
#define NROW  8192     // B*S
#define DHEAD 64
#define NHEAD 8

__device__ __forceinline__ v8f wmma16(v16h a, v16h b, v8f c) {
  // D = A(16x32 f16) x B(32x16 f16) + C(16x16 f32)
  return __builtin_amdgcn_wmma_f32_16x16x32_f16(false, a, false, b, (short)0, c,
                                                false, false);
}

// ---------------------------------------------------------------------------
// P = X[8192x128] @ W[512x128]^T  -> f16.  mode 0: row-major [8192][512]
// mode 1 (V): store transposed per head: Vt[b][h][d][s]
// ---------------------------------------------------------------------------
__global__ __launch_bounds__(256) void proj_kernel(const float* __restrict__ X,
                                                   const float* __restrict__ W,
                                                   _Float16* __restrict__ out,
                                                   int mode) {
  const int lane = threadIdx.x & 31;
  const int warp = threadIdx.x >> 5;
  const int g    = blockIdx.x * 8 + warp;   // 16384 wave tiles
  const int rt   = g >> 5;                  // 512 row tiles
  const int ct   = g & 31;                  // 32 col tiles
  const int lo   = lane & 15;
  const int hi8  = (lane & 16) ? 8 : 0;
  const int hi16 = (lane & 16) ? 16 : 0;

  const float* xrow = X + (size_t)(rt * 16 + lo) * EMB;
  const float* wrow = W + (size_t)(ct * 16 + lo) * EMB;

  v8f c = {};
#pragma unroll
  for (int kc = 0; kc < 4; ++kc) {
    const int k0 = kc * 32;
    v16h a, b;
#pragma unroll
    for (int i = 0; i < 16; i += 2) {  // A-layout: contiguous K per lane half
      const int ka = k0 + hi16 + i;
      const float2 f2 = *(const float2*)(xrow + ka);
      a[i]     = (_Float16)f2.x;
      a[i + 1] = (_Float16)f2.y;
    }
#pragma unroll
    for (int i = 0; i < 16; i += 4) {  // B-layout: contiguous K per lane half
      const float4 f4 = *(const float4*)(wrow + k0 + hi16 + i);
      b[i]     = (_Float16)f4.x;
      b[i + 1] = (_Float16)f4.y;
      b[i + 2] = (_Float16)f4.z;
      b[i + 3] = (_Float16)f4.w;
    }
    c = wmma16(a, b, c);
  }

#pragma unroll
  for (int j = 0; j < 8; ++j) {   // D layout: row = hi8+j, col = lane&15
    const int r   = rt * 16 + hi8 + j;
    const int col = ct * 16 + lo;
    const _Float16 v = (_Float16)c[j];
    if (mode == 0) {
      out[(size_t)r * HDIM + col] = v;
    } else {
      const int bb = r >> 11;                    // batch
      const int f  = ((r & 2047) << 9) + col;    // flat index within batch
      const int h  = f >> 17;
      const int q  = (f >> 6) & 2047;
      const int d  = f & 63;
      out[((size_t)((bb * NHEAD + h) * DHEAD + d)) * SEQ + q] = v;
    }
  }
}

// ---------------------------------------------------------------------------
// Fused flash attention per (b,h): scores -> mask(1e-10) -> online softmax -> PV
// 1 wave = 16 query rows; block = 8 waves = 128 rows; grid = 32 heads * 16.
// ---------------------------------------------------------------------------
__global__ __launch_bounds__(256) void attn_kernel(const _Float16* __restrict__ Qh,
                                                   const _Float16* __restrict__ Kh,
                                                   const _Float16* __restrict__ Vt,
                                                   const unsigned char* __restrict__ mask,
                                                   _Float16* __restrict__ Zb) {
  __shared__ _Float16 pbuf[8][16][34];  // per-wave 16x32 P tile (padded)
  const int lane = threadIdx.x & 31;
  const int warp = threadIdx.x >> 5;
  const int lo   = lane & 15;
  const int hi8  = (lane & 16) ? 8 : 0;
  const int hi16 = (lane & 16) ? 16 : 0;
  const int bh   = blockIdx.x >> 4;  // 0..31
  const int qt   = blockIdx.x & 15;
  const int b    = bh >> 3;
  const size_t headOff = (size_t)bh * SEQ * DHEAD;
  const _Float16* Qp = Qh + headOff;
  const _Float16* Kp = Kh + headOff;
  const _Float16* Vp = Vt + headOff;
  const unsigned char* mp = mask + (size_t)b * SEQ * SEQ;
  const int qbase = qt * 128 + warp * 16;

  // Load the wave's Q tile as two A-operands (d 0..31 and d 32..63)
  v16h qa0, qa1;
  {
    const _Float16* qrow = Qp + (size_t)(qbase + lo) * DHEAD;
#pragma unroll
    for (int i = 0; i < 16; i += 2) {
      const int d = hi16 + i;
      const v2h p0 = *(const v2h*)(qrow + d);
      const v2h p1 = *(const v2h*)(qrow + 32 + d);
      qa0[i] = p0[0]; qa0[i + 1] = p0[1];
      qa1[i] = p1[0]; qa1[i + 1] = p1[1];
    }
  }

  float M[8], L[8];
#pragma unroll
  for (int j = 0; j < 8; ++j) { M[j] = -1e30f; L[j] = 0.f; }
  v8f O0 = {}, O1 = {}, O2 = {}, O3 = {};

  for (int kc = 0; kc < SEQ; kc += 32) {
    if (kc + 32 < SEQ)
      __builtin_prefetch(Kp + (size_t)(kc + 32 + lo) * DHEAD, 0, 1);

    // ---- scores: two 16x16 tiles (keys kc..+15, kc+16..+31) ----
    v8f c0 = {}, c1 = {};
    {
      const _Float16* kr0 = Kp + (size_t)(kc + lo) * DHEAD + hi16;
      const _Float16* kr1 = Kp + (size_t)(kc + 16 + lo) * DHEAD + hi16;
      c0 = wmma16(qa0, *(const v16h*)(kr0), c0);
      c0 = wmma16(qa1, *(const v16h*)(kr0 + 32), c0);
      c1 = wmma16(qa0, *(const v16h*)(kr1), c1);
      c1 = wmma16(qa1, *(const v16h*)(kr1 + 32), c1);
    }

    // ---- masked fill 1e-10 (applied before softmax, like the reference) ----
    float s0[8], s1[8];
    const int key0 = kc + lo;
#pragma unroll
    for (int j = 0; j < 8; ++j) {
      const unsigned char* mrow = mp + (size_t)(qbase + hi8 + j) * SEQ;
      float v0 = c0[j]; if (mrow[key0])      v0 = 1e-10f;
      float v1 = c1[j]; if (mrow[key0 + 16]) v1 = 1e-10f;
      s0[j] = v0; s1[j] = v1;
    }

    // ---- online softmax (row = hi8+j, replicated across 16-lane group) ----
#pragma unroll
    for (int j = 0; j < 8; ++j) {
      float rm = fmaxf(s0[j], s1[j]);
      rm = fmaxf(rm, __shfl_xor(rm, 1, 16));
      rm = fmaxf(rm, __shfl_xor(rm, 2, 16));
      rm = fmaxf(rm, __shfl_xor(rm, 4, 16));
      rm = fmaxf(rm, __shfl_xor(rm, 8, 16));
      const float mn = fmaxf(M[j], rm);
      const float sc = __expf(M[j] - mn);
      M[j] = mn;
      const float p0 = __expf(s0[j] - mn);
      const float p1 = __expf(s1[j] - mn);
      float rs = p0 + p1;
      rs += __shfl_xor(rs, 1, 16);
      rs += __shfl_xor(rs, 2, 16);
      rs += __shfl_xor(rs, 4, 16);
      rs += __shfl_xor(rs, 8, 16);
      L[j] = L[j] * sc + rs;
      O0[j] *= sc; O1[j] *= sc; O2[j] *= sc; O3[j] *= sc;
      s0[j] = p0; s1[j] = p1;
    }

    // ---- transpose P (D-layout -> A-layout) through LDS ----
#pragma unroll
    for (int j = 0; j < 8; ++j) {
      pbuf[warp][hi8 + j][lo]      = (_Float16)s0[j];
      pbuf[warp][hi8 + j][16 + lo] = (_Float16)s1[j];
    }
    asm volatile("s_wait_dscnt 0" ::: "memory");
    v16h pa;
#pragma unroll
    for (int i = 0; i < 16; ++i) {
      const int k = hi16 + i;
      pa[i] = pbuf[warp][lo][k];
    }
    asm volatile("" ::: "memory");

    // ---- O += P(16x32) x V(32x16) per d-chunk; Vt rows are contiguous ----
    const _Float16* vb = Vp + kc + hi16;
    O0 = wmma16(pa, *(const v16h*)(vb + (size_t)(lo)      * SEQ), O0);
    O1 = wmma16(pa, *(const v16h*)(vb + (size_t)(16 + lo) * SEQ), O1);
    O2 = wmma16(pa, *(const v16h*)(vb + (size_t)(32 + lo) * SEQ), O2);
    O3 = wmma16(pa, *(const v16h*)(vb + (size_t)(48 + lo) * SEQ), O3);
  }

  // ---- normalize and store Z (f16, contiguous [B][H][S][D]) ----
#pragma unroll
  for (int j = 0; j < 8; ++j) {
    const float inv = 1.0f / L[j];
    _Float16* zr = Zb + headOff + (size_t)(qbase + hi8 + j) * DHEAD + lo;
    zr[0]  = (_Float16)(O0[j] * inv);
    zr[16] = (_Float16)(O1[j] * inv);
    zr[32] = (_Float16)(O2[j] * inv);
    zr[48] = (_Float16)(O3[j] * inv);
  }
}

// ---------------------------------------------------------------------------
// Y = LayerNorm( Z[8192x512] @ Wfc[128x512]^T + inputQ )
// block = 16 rows x 128 cols (8 waves, one 16x16 tile each), then LN in-block.
// ---------------------------------------------------------------------------
__global__ __launch_bounds__(256) void out_kernel(const _Float16* __restrict__ Zb,
                                                  const float* __restrict__ Wfc,
                                                  const float* __restrict__ Xq,
                                                  float* __restrict__ Y) {
  __shared__ float ybuf[16][132];
  const int lane = threadIdx.x & 31;
  const int warp = threadIdx.x >> 5;
  const int lo   = lane & 15;
  const int hi8  = (lane & 16) ? 8 : 0;
  const int hi16 = (lane & 16) ? 16 : 0;
  const int rt   = blockIdx.x;  // 512 row tiles
  const int ct   = warp;        // 8 col tiles

  const _Float16* zrow = Zb + (size_t)(rt * 16 + lo) * HDIM;
  const float*    wrow = Wfc + (size_t)(ct * 16 + lo) * HDIM;
  v8f c = {};
#pragma unroll
  for (int kc = 0; kc < 16; ++kc) {
    const int k0 = kc * 32;
    v16h a, b;
#pragma unroll
    for (int i = 0; i < 16; i += 2) {
      const int ka = k0 + hi16 + i;
      const v2h p = *(const v2h*)(zrow + ka);
      a[i] = p[0]; a[i + 1] = p[1];
    }
#pragma unroll
    for (int i = 0; i < 16; i += 4) {
      const float4 f4 = *(const float4*)(wrow + k0 + hi16 + i);
      b[i]     = (_Float16)f4.x;
      b[i + 1] = (_Float16)f4.y;
      b[i + 2] = (_Float16)f4.z;
      b[i + 3] = (_Float16)f4.w;
    }
    c = wmma16(a, b, c);
  }
#pragma unroll
  for (int j = 0; j < 8; ++j) {
    const int r   = rt * 16 + hi8 + j;
    const int col = ct * 16 + lo;
    ybuf[hi8 + j][col] = c[j] + Xq[(size_t)r * EMB + col];
  }
  __syncthreads();

  // LayerNorm: 16 threads per row, 8 cols each
  const int row = threadIdx.x >> 4;
  const int cb  = (threadIdx.x & 15) * 8;
  float s = 0.f, s2 = 0.f;
#pragma unroll
  for (int i = 0; i < 8; ++i) {
    const float v = ybuf[row][cb + i];
    s += v; s2 += v * v;
  }
  s += __shfl_xor(s, 1, 16);  s2 += __shfl_xor(s2, 1, 16);
  s += __shfl_xor(s, 2, 16);  s2 += __shfl_xor(s2, 2, 16);
  s += __shfl_xor(s, 4, 16);  s2 += __shfl_xor(s2, 4, 16);
  s += __shfl_xor(s, 8, 16);  s2 += __shfl_xor(s2, 8, 16);
  const float mean = s * (1.0f / 128.0f);
  const float var  = s2 * (1.0f / 128.0f) - mean * mean;
  const float rstd = rsqrtf(var + 1e-5f);
  const int r = rt * 16 + row;
#pragma unroll
  for (int i = 0; i < 8; ++i)
    Y[(size_t)r * EMB + cb + i] = (ybuf[row][cb + i] - mean) * rstd;
}

// ---------------------------------------------------------------------------
extern "C" void kernel_launch(void* const* d_in, const int* in_sizes, int n_in,
                              void* d_out, int out_size, void* d_ws, size_t ws_size,
                              hipStream_t stream) {
  (void)in_sizes; (void)n_in; (void)out_size; (void)ws_size;
  const float* inQ = (const float*)d_in[0];
  const float* inK = (const float*)d_in[1];
  const float* inV = (const float*)d_in[2];
  const float* WQ  = (const float*)d_in[3];
  const float* Wfc = (const float*)d_in[4];
  const unsigned char* mask = (const unsigned char*)d_in[5];
  float* out = (float*)d_out;

  // Workspace: 4 x 8 MiB f16 buffers (Qh, Kh, Vt, Z) = 32 MiB
  _Float16* qh = (_Float16*)d_ws;
  _Float16* kh = qh + (size_t)NROW * HDIM;
  _Float16* vt = kh + (size_t)NROW * HDIM;
  _Float16* zb = vt + (size_t)NROW * HDIM;

  proj_kernel<<<2048, 256, 0, stream>>>(inQ, WQ, qh, 0);
  proj_kernel<<<2048, 256, 0, stream>>>(inK, WQ, kh, 0);
  proj_kernel<<<2048, 256, 0, stream>>>(inV, WQ, vt, 1);
  attn_kernel<<<512, 256, 0, stream>>>(qh, kh, vt, mask, zb);
  out_kernel<<<512, 256, 0, stream>>>(zb, Wfc, inQ, out);
}
